// VIN_62715112456540
// MI455X (gfx1250) — matrix operand order repaired and checked by reading, not compiled, not run on hardware
//
#include <hip/hip_runtime.h>
#include <hip/hip_bf16.h>

// ---------------------------------------------------------------------------
// VIN fused kernel for gfx1250 (MI455X), wave32, WMMA f16 16x16x32.
//
//  - h/r convs collapsed via composed kernel W2 (162 floats): 0.26 GFLOP and
//    zero HBM traffic instead of 6.5 GFLOP + 482 MB intermediate.
//  - One workgroup (256 thr = 8 waves) per image; r and ping-pong v buffers
//    live in LDS (30x30, zero halo) for all 51 sequential iterations.
//  - q-conv per iteration = GEMM; orientation: M=channels (A = q-weights,
//    loop-invariant registers), N=pixels (B = im2col patches). D then holds
//    channels 0-7 of a pixel in one lane and 8-9 in its +16 partner, so the
//    channel-max is 8 in-register v_max + ONE shfl_xor(16) + one ds_store
//    (previous version burned 32 wait-serialized ds_bpermute per tile).
//  - r scaled by 4096 (iteration map is positively homogeneous, v0 = 0) so
//    f16 operands stay in normal range; output divided once at the end.
// ---------------------------------------------------------------------------

typedef __attribute__((ext_vector_type(16))) _Float16 v16h;
typedef __attribute__((ext_vector_type(8)))  float    v8f;

union AV { v16h v; _Float16 h[16]; };

#define IMS   28
#define NPIX  (IMS * IMS)      // 784
#define HW    30               // halo'd tile edge (1-px zero border)
#define XSW   32               // X tile edge (2-px zero border)
#define CH_H  150
#define KITER 51               // K + 1
#define SCALE 4096.0f
#define INV_SCALE (1.0f / SCALE)

__global__ __launch_bounds__(256) void vin_fused_kernel(
    const float* __restrict__ total,   // [B, 1570]
    const float* __restrict__ h1_k,    // [3,3,2,150]
    const float* __restrict__ h1_b,    // [150]
    const float* __restrict__ r1_k,    // [3,3,150,1]
    const float* __restrict__ q1_k,    // [3,3,2,10]
    float* __restrict__ out,           // [B, 784]
    int B)
{
    __shared__ float Xs[XSW * XSW * 2];   // X with 2-halo (zeros)
    __shared__ float w2[162];             // composed h∘r kernel
    __shared__ float bt[9];               // composed bias term
    __shared__ float rh[HW * HW];         // r (scaled), 1-halo
    __shared__ float vb0[HW * HW];        // v ping
    __shared__ float vb1[HW * HW];        // v pong
    __shared__ float h1s[9 * 2 * CH_H];   // staged h1_k
    __shared__ float r1s[9 * CH_H];       // staged r1_k

    const int b    = blockIdx.x;
    const int tid  = threadIdx.x;
    const int lane = tid & 31;            // wave32
    const int wid  = tid >> 5;            // 8 waves
    const int half = lane >> 4;           // 0: lanes 0-15, 1: lanes 16-31
    const int col  = lane & 15;           // pixel-in-tile (N) / channel row

    const float* tot = total + (size_t)b * 1570 + 2;
    __builtin_prefetch(tot, 0, 3);        // global_prefetch_b8

    // ---- Phase 0: stage X (zero halo) and the big weights into LDS --------
    for (int i = tid; i < XSW * XSW * 2; i += 256) Xs[i] = 0.0f;
    for (int i = tid; i < HW * HW; i += 256) { vb0[i] = 0.0f; vb1[i] = 0.0f; rh[i] = 0.0f; }
    for (int i = tid; i < 9 * 2 * CH_H; i += 256) h1s[i] = h1_k[i];
    for (int i = tid; i < 9 * CH_H;     i += 256) r1s[i] = r1_k[i];
    for (int i = tid; i < NPIX * 2; i += 256) {
        int ci = i & 1, p = i >> 1;
        int y = p / IMS, x = p % IMS;
        Xs[((y + 2) * XSW + (x + 2)) * 2 + ci] = tot[i];
    }
    __syncthreads();

    // ---- Phase 1: composed kernel  W2[d1][d2][ci] = sum_c r1*h1 -----------
    if (tid < 162) {
        int d1 = tid / 18, rest = tid % 18, d2 = rest >> 1, ci = rest & 1;
        float acc = 0.0f;
        for (int c = 0; c < CH_H; ++c)
            acc += r1s[d1 * CH_H + c] * h1s[(d2 * 2 + ci) * CH_H + c];
        w2[tid] = acc;
    } else if (tid < 171) {
        int d1 = tid - 162;
        float acc = 0.0f;
        for (int c = 0; c < CH_H; ++c) acc += r1s[d1 * CH_H + c] * h1_b[c];
        bt[d1] = acc;
    }
    __syncthreads();

    // ---- Phase 2: r = (conv2 ∘ conv1)(X), boundary-exact, scaled ----------
    for (int p = tid; p < NPIX; p += 256) {
        int y = p / IMS, x = p % IMS;
        float acc = 0.0f;
        #pragma unroll
        for (int ry = 0; ry < 3; ++ry) {
            #pragma unroll
            for (int rx = 0; rx < 3; ++rx) {
                int Y = y + ry - 1, X = x + rx - 1;
                if ((unsigned)Y < (unsigned)IMS && (unsigned)X < (unsigned)IMS) {
                    int d1 = ry * 3 + rx;
                    float a2 = bt[d1];
                    #pragma unroll
                    for (int d2 = 0; d2 < 9; ++d2) {
                        int ky = d2 / 3, kx = d2 % 3;
                        int xi = ((Y + ky + 1) * XSW + (X + kx + 1)) * 2;
                        a2 += w2[d1 * 18 + d2 * 2 + 0] * Xs[xi + 0];
                        a2 += w2[d1 * 18 + d2 * 2 + 1] * Xs[xi + 1];
                    }
                    acc += a2;
                }
            }
        }
        rh[(y + 1) * HW + (x + 1)] = acc * SCALE;
    }

    // ---- A operand (loop invariant): channels on M. A[M=co][K] ------------
    // 16-bit A 16x32 layout: lanes 0-15 AND 16-31 both carry M = lane%16;
// element e -> K = 16*half + e (contiguous, matches B).
    // Wq[K][co] = q1_k[K*10 + co]; zero-pad K>=18, co>=10.
    AV aw;
    #pragma unroll
    for (int e = 0; e < 16; ++e) {
        int K = 16 * half + e;
        float w = (K < 18 && col < 10) ? q1_k[K * 10 + col] : 0.0f;
        aw.h[e] = (_Float16)w;
    }
    __syncthreads();

    // ---- Phase 3: 51 value-iteration steps, all in LDS + WMMA -------------
    for (int it = 0; it < KITER; ++it) {
        const float* vin  = (it & 1) ? vb1 : vb0;
        float*       vout = (it & 1) ? vb0 : vb1;

        for (int tile = wid; tile < 49; tile += 8) {   // uniform per wave
            const int p    = tile * 16 + col;          // this lane's pixel
            const int y    = p / IMS;                  // one div per tile
            const int base = p + 2 * y + (HW + 1);     // (y+1)*30 + (x+1)

            // B operand (patches): 32x16 layout, lane holds column N=col,
            // element e -> K = 16*half + e. K = tap*2 + ch (ch0=r, ch1=v).
            // Low half: K=0..15, all offsets compile-time. High half: only
            // K=16 (tap8,r) and K=17 (tap8,v) are real; tap8 -> (+1,+1).
            AV bv;
            if (half == 0) {
                #pragma unroll
                for (int e = 0; e < 16; ++e) {
                    const int tap = e >> 1;
                    const int off = (tap / 3 - 1) * HW + (tap % 3 - 1);
                    const float* s = (e & 1) ? vin : rh;
                    bv.h[e] = (_Float16)s[base + off];
                }
            } else {
                bv.h[0] = (_Float16)rh[base + HW + 1];
                bv.h[1] = (_Float16)vin[base + HW + 1];
                #pragma unroll
                for (int e = 2; e < 16; ++e) bv.h[e] = (_Float16)0.0f;
            }

            // D[ch][px] (fp32): lane L<16 gets ch 0..7 of pixel L in d[0..7];
            // lane L+16 gets ch 8..15 (only 8,9 real) of the same pixel.
            v8f acc = {};
            v8f d = __builtin_amdgcn_wmma_f32_16x16x32_f16(
                false, aw.v, false, bv.v, (short)0, acc, false, false);

            // Channel max: in-register tree + one cross-half exchange.
            float m01 = fmaxf(d[0], d[1]);
            float m27 = fmaxf(fmaxf(fmaxf(d[2], d[3]), fmaxf(d[4], d[5])),
                              fmaxf(d[6], d[7]));
            float m = half ? m01 : fmaxf(m01, m27);
            m = fmaxf(m, __shfl_xor(m, 16, 32));
            if (half == 0) vout[base] = m;
        }
        __syncthreads();
    }

    // ---- Writeout: iteration 50 (even) wrote vb1; undo the scaling --------
    for (int p = tid; p < NPIX; p += 256) {
        int y = p / IMS, x = p % IMS;
        out[(size_t)b * NPIX + p] = vb1[(y + 1) * HW + (x + 1)] * INV_SCALE;
    }
}

extern "C" void kernel_launch(void* const* d_in, const int* in_sizes, int n_in,
                              void* d_out, int out_size, void* d_ws, size_t ws_size,
                              hipStream_t stream) {
    (void)n_in; (void)out_size; (void)d_ws; (void)ws_size;
    const float* total = (const float*)d_in[0];
    const float* h1_k  = (const float*)d_in[1];
    const float* h1_b  = (const float*)d_in[2];
    const float* r1_k  = (const float*)d_in[3];
    const float* q1_k  = (const float*)d_in[4];
    float* out = (float*)d_out;
    const int B = in_sizes[0] / 1570;
    vin_fused_kernel<<<dim3(B), dim3(256), 0, stream>>>(
        total, h1_k, h1_b, r1_k, q1_k, out, B);
}
